// SelfAttn_54709293416927
// MI455X (gfx1250) — matrix operand order repaired and checked by reading, not compiled, not run on hardware
//
#include <hip/hip_runtime.h>
#include <hip/hip_bf16.h>

typedef __bf16 bf16;
typedef __attribute__((ext_vector_type(8)))  __bf16 v8bf;
typedef __attribute__((ext_vector_type(16))) __bf16 v16bf;
typedef __attribute__((ext_vector_type(8)))  float  v8f;

#define BB 4
#define NQ 4096
#define HH 512
#define DD 256

static __device__ __forceinline__ v16bf make_v16(v8bf lo, v8bf hi) {
    return __builtin_shufflevector(lo, hi, 0,1,2,3,4,5,6,7,8,9,10,11,12,13,14,15);
}

// A fragment: 16x32 bf16 tile, row-major source with leading dim `ld`.
// lanes 0-15  (row = lane):    elems 0-7 = K+0..7,  elems 8-15 = K+16..23
// lanes 16-31 (row = lane-16): elems 0-7 = K+8..15, elems 8-15 = K+24..31
static __device__ __forceinline__ v16bf load_afrag(const bf16* p0, int ld, int lane) {
    int l = lane & 15, h = lane >> 4;
    const bf16* p = p0 + l * ld + h * 8;
    v8bf lo = *(const v8bf*)(p);
    v8bf hi = *(const v8bf*)(p + 16);
    return make_v16(lo, hi);
}

// B fragment: 32x16 bf16, where column n of B is row n of a row-major matrix Y
// (i.e. B = Y^T). lanes 0-15: col = lane, K+0..7, K+16..23; lanes 16-31: col = lane-16, K+8..15, K+24..31.
static __device__ __forceinline__ v16bf load_bfrag(const bf16* p0, int ld, int lane) {
    int l = lane & 15, h = lane >> 4;
    const bf16* p = p0 + l * ld + h * 8;
    v8bf lo = *(const v8bf*)(p);
    v8bf hi = *(const v8bf*)(p + 16);
    return make_v16(lo, hi);
}

static __device__ __forceinline__ v8f wmma_bf16(v16bf a, v16bf b, v8f c) {
    return __builtin_amdgcn_wmma_f32_16x16x32_bf16(false, a, false, b, (short)0, c, false, false);
}

// ---------------- Kernel 0: W2 [H,D] f32 -> W2T [D,H] bf16 ----------------
__global__ __launch_bounds__(256) void k_w2t(const float* __restrict__ W2,
                                             bf16* __restrict__ W2T) {
    int idx = blockIdx.x * 256 + threadIdx.x;    // < HH*DD
    int h = idx >> 8;                            // / DD
    int d = idx & (DD - 1);
    W2T[(size_t)d * HH + h] = (bf16)W2[idx];
}

// ---------------- Kernel 1: INR + scale -> Qv/Kv bf16 ----------------
// Block: 64 rows of (B*N). 8 waves: wave w -> rows 16*(w&3), cols 128*(w>>2).
__global__ __launch_bounds__(256) void k_inr(
    const float* __restrict__ Qc, const float* __restrict__ Kc,
    const float* __restrict__ W1, const float* __restrict__ b1,
    const bf16*  __restrict__ W2T, const float* __restrict__ b2,
    const float* __restrict__ Qw, const float* __restrict__ Kw,
    bf16* __restrict__ Qv, bf16* __restrict__ Kv)
{
    const int t   = threadIdx.x;
    const int isK = blockIdx.y;
    const int r0  = blockIdx.x * 64;             // row offset in [0, B*N)
    const float* C = isK ? Kc : Qc;
    const float* W = isK ? Kw : Qw;
    bf16*        V = isK ? Kv : Qv;

    __shared__ bf16 Hs[64 * 520];                // 520 = 512 + 8 pad (bank skew, 16B aligned pitch)

    // Stage 1: h = tanh(coords @ W1 + b1), stored bf16 in LDS.
    #pragma unroll 4
    for (int i = 0; i < 128; ++i) {
        int idx = i * 256 + t;                   // 0..64*512-1
        int row = idx >> 9;
        int j   = idx & (HH - 1);
        float c0 = C[(size_t)(r0 + row) * 2 + 0];
        float c1 = C[(size_t)(r0 + row) * 2 + 1];
        float hv = tanhf(fmaf(c0, W1[j], fmaf(c1, W1[HH + j], b1[j])));
        Hs[row * 520 + j] = (bf16)hv;
    }
    __syncthreads();

    // Stage 2: out[64,256] = h[64,512] @ W2 (B-frags from W2T, contiguous in K).
    const int w = t >> 5, lane = t & 31;
    const int wm = w & 3;                        // 16-row group
    const int wn = w >> 2;                       // 128-col group
    v8f acc[8] = {};
    const bf16* hbase = &Hs[(16 * wm) * 520];
    for (int k0 = 0; k0 < HH; k0 += 32) {
        v16bf a = load_afrag(hbase + k0, 520, lane);
        #pragma unroll
        for (int nt = 0; nt < 8; ++nt) {
            const bf16* bb = W2T + (size_t)(128 * wn + 16 * nt) * HH + k0;
            v16bf bfr = load_bfrag(bb, HH, lane);
            acc[nt] = wmma_bf16(a, bfr, acc[nt]);
        }
    }

    // Epilogue: + b2, * weights[n], -> bf16.
    const int l = lane & 15, half = lane >> 4;
    #pragma unroll
    for (int nt = 0; nt < 8; ++nt) {
        int col = 128 * wn + 16 * nt + l;
        float bias = b2[col];
        #pragma unroll
        for (int i = 0; i < 8; ++i) {
            int r = r0 + 16 * wm + 8 * half + i;
            int n = r & (NQ - 1);
            float val = (acc[nt][i] + bias) * W[(size_t)n * DD + col];
            V[(size_t)r * DD + col] = (bf16)val;
        }
    }
}

// ---------------- Kernel 2: QK[b] = Qv[b] (N x D) @ Kv[b]^T ----------------
// Block: 128x128 output tile, 8 waves; K chunked by 128 through LDS.
__global__ __launch_bounds__(256) void k_qk(const bf16* __restrict__ Qv,
                                            const bf16* __restrict__ Kv,
                                            float* __restrict__ out)
{
    const int t   = threadIdx.x;
    const int bn0 = blockIdx.x * 128;
    const int bm0 = blockIdx.y * 128;
    const int b   = blockIdx.z;
    const bf16* Aq = Qv + (size_t)b * NQ * DD;
    const bf16* Bk = Kv + (size_t)b * NQ * DD;

    __shared__ bf16 As[128 * 136];               // 136 = 128 + 8 pad
    __shared__ bf16 Bs[128 * 136];

    const int w = t >> 5, lane = t & 31;
    const int wm = w & 3;                        // 32-row group
    const int wn = w >> 2;                       // 64-col group

    v8f acc[2][4] = {};

    for (int kk = 0; kk < DD; kk += 128) {
        #pragma unroll
        for (int i = 0; i < 8; ++i) {
            int c = i * 256 + t;                 // 0..2047 chunks of 8 bf16
            int row  = c >> 4;
            int col8 = (c & 15) << 3;
            *(v8bf*)(As + row * 136 + col8) =
                *(const v8bf*)(Aq + (size_t)(bm0 + row) * DD + kk + col8);
            *(v8bf*)(Bs + row * 136 + col8) =
                *(const v8bf*)(Bk + (size_t)(bn0 + row) * DD + kk + col8);
        }
        __syncthreads();

        #pragma unroll
        for (int k0 = 0; k0 < 128; k0 += 32) {
            v16bf a0 = load_afrag(As + (32 * wm +  0) * 136 + k0, 136, lane);
            v16bf a1 = load_afrag(As + (32 * wm + 16) * 136 + k0, 136, lane);
            #pragma unroll
            for (int nt = 0; nt < 4; ++nt) {
                v16bf bfr = load_bfrag(Bs + (64 * wn + 16 * nt) * 136 + k0, 136, lane);
                acc[0][nt] = wmma_bf16(a0, bfr, acc[0][nt]);
                acc[1][nt] = wmma_bf16(a1, bfr, acc[1][nt]);
            }
        }
        __syncthreads();
    }

    const int l = lane & 15, half = lane >> 4;
    #pragma unroll
    for (int mi = 0; mi < 2; ++mi) {
        #pragma unroll
        for (int nt = 0; nt < 4; ++nt) {
            int col = bn0 + 64 * wn + 16 * nt + l;
            size_t base = ((size_t)b * NQ + bm0 + 32 * wm + 16 * mi + 8 * half) * (size_t)NQ + col;
            #pragma unroll
            for (int i = 0; i < 8; ++i)
                out[base + (size_t)i * NQ] = acc[mi][nt][i];
        }
    }
}

extern "C" void kernel_launch(void* const* d_in, const int* in_sizes, int n_in,
                              void* d_out, int out_size, void* d_ws, size_t ws_size,
                              hipStream_t stream) {
    const float* Qc = (const float*)d_in[0];
    const float* Kc = (const float*)d_in[1];
    const float* W1 = (const float*)d_in[2];
    const float* b1 = (const float*)d_in[3];
    const float* W2 = (const float*)d_in[4];
    const float* b2 = (const float*)d_in[5];
    const float* Qw = (const float*)d_in[6];
    const float* Kw = (const float*)d_in[7];

    // Workspace layout (requires ~17 MB):
    //   [0, 256KB)  : W2T bf16  (D x H)
    //   [1MB, 9MB)  : Qv  bf16  (B*N x D)
    //   [9MB, 17MB) : Kv  bf16  (B*N x D)
    char* ws  = (char*)d_ws;
    bf16* W2T = (bf16*)(ws);
    bf16* Qv  = (bf16*)(ws + (size_t)(1 << 20));
    bf16* Kv  = (bf16*)(ws + (size_t)(9 << 20));

    k_w2t<<<dim3((HH * DD) / 256), dim3(256), 0, stream>>>(W2, W2T);
    k_inr<<<dim3((BB * NQ) / 64, 2), dim3(256), 0, stream>>>(
        Qc, Kc, W1, b1, W2T, b2, Qw, Kw, Qv, Kv);
    k_qk<<<dim3(NQ / 128, NQ / 128, BB), dim3(256), 0, stream>>>(
        Qv, Kv, (float*)d_out);
}
